// PressGNN_45792941310332
// MI455X (gfx1250) — compile-verified, matrix operand order checked by reading, not run on hardware
//
#include <hip/hip_runtime.h>

// f16 vector types for CDNA5 WMMA
typedef _Float16 half_t;
typedef __attribute__((ext_vector_type(16))) _Float16 v16h;
typedef __attribute__((ext_vector_type(8)))  float    v8f;

#define NODES   22
#define ROWP    32        // node rows padded to 2 WMMA M-tiles
#define LDH     512       // LDS row stride (halfs)
#define FIN     32
#define D1      512       // H1*C1
#define C1      128
#define NTHREADS 256      // 8 wave32

// Packed-weight sizes (in halfs): [ntiles][ksteps][lane(32)][elem(16)]
#define W1P_ELEMS (32 * 1  * 512)   // K=32,  N=512
#define W2P_ELEMS (32 * 16 * 512)   // K=512, N=512
#define W3P_ELEMS (8  * 16 * 512)   // K=512, N=128

// K index inside a 16x32 f16 WMMA A/B fragment for element i of a lane
// (ISA 7.12.2: v0..3 hold K 0..7 / 8..15 split by lane-half, v4..7 hold +16)
__device__ __forceinline__ int kidx(int i, int lane) {
  return ((i & 8) << 1) + ((lane >> 4) << 3) + (i & 7);
}

// ---------------- weight pre-pack (f32 -> f16 in B-fragment order) ---------
__device__ __forceinline__ void pack_one(int idx, const float* __restrict__ W,
                                         int K, int N, half_t* __restrict__ out) {
  const int ksteps = K >> 5;
  const int per    = ksteps << 9;          // ksteps * 512 elems per ntile
  int ntile = idx / per;
  int r     = idx - ntile * per;
  int kstep = r >> 9;
  int r2    = r & 511;
  int lane  = r2 >> 4;
  int i     = r2 & 15;
  int k = (kstep << 5) + kidx(i, lane);
  int n = (ntile << 4) + (lane & 15);
  out[idx] = (half_t)W[(size_t)k * N + n];
}

__global__ __launch_bounds__(NTHREADS) void pack_weights(
    const float* __restrict__ W1, const float* __restrict__ W2,
    const float* __restrict__ W3, half_t* __restrict__ ws) {
  int idx = blockIdx.x * blockDim.x + threadIdx.x;
  if (idx < W1P_ELEMS) {
    pack_one(idx, W1, FIN, D1, ws);
  } else if (idx < W1P_ELEMS + W2P_ELEMS) {
    pack_one(idx - W1P_ELEMS, W2, D1, D1, ws + W1P_ELEMS);
  } else if (idx < W1P_ELEMS + W2P_ELEMS + W3P_ELEMS) {
    pack_one(idx - W1P_ELEMS - W2P_ELEMS, W3, D1, C1, ws + W1P_ELEMS + W2P_ELEMS);
  }
}

// ---------------- H = X @ W   (X in LDS f16, W pre-packed f16) -------------
// Register-blocked: one A fragment (from LDS) feeds NB consecutive N-tiles,
// NB independent accumulators -> no D->A/B WMMA hazards, 4x fewer LDS reads.
template <int NB>
__device__ void mm_stage(const half_t* __restrict__ Wp,
                         const half_t* __restrict__ X, half_t* __restrict__ H,
                         int K, int N, int wave, int lane) {
  const int ksteps  = K >> 5;
  const int ngroups = (N >> 4) / NB;
  const int units   = 2 * ngroups;         // 2 M-tiles
  const int lo = lane & 15;
  const int hi = lane >> 4;
  for (int u = wave; u < units; u += 8) {
    int mtile = u / ngroups;
    int ng    = u - mtile * ngroups;
    int ntile0 = ng * NB;
    v8f acc[NB];
    const v8f vzero = {};
#pragma unroll
    for (int b = 0; b < NB; ++b) acc[b] = vzero;
    const int mrow = (mtile * 16 + lo) * LDH;
    for (int ks = 0; ks < ksteps; ++ks) {
      v16h a;
      const int kbase = mrow + (ks << 5);
#pragma unroll
      for (int i = 0; i < 16; ++i) a[i] = X[kbase + kidx(i, lane)];
#pragma unroll
      for (int b = 0; b < NB; ++b) {
        const half_t* bp =
            Wp + ((((size_t)(ntile0 + b) * ksteps + ks) << 5) + lane) * 16;
        v16h bw = *(const v16h*)bp;
        acc[b] = __builtin_amdgcn_wmma_f32_16x16x32_f16(
            false, a, false, bw, (short)0, acc[b], false, false);
      }
    }
#pragma unroll
    for (int b = 0; b < NB; ++b) {
      const int nbase = (ntile0 + b) * 16 + lo;
#pragma unroll
      for (int v = 0; v < 8; ++v) {
        int m = mtile * 16 + (hi << 3) + v;
        H[m * LDH + nbase] = (half_t)acc[b][v];
      }
    }
  }
}

// ---------------- GAT attention: scores, softmax, alpha@H + bias + relu ----
__device__ void gat_attention(const half_t* __restrict__ Hs, half_t* __restrict__ O,
                              half_t* __restrict__ Al,
                              float* __restrict__ s_sh, float* __restrict__ d_sh,
                              const float* __restrict__ asv, const float* __restrict__ adv,
                              const float* __restrict__ bias,
                              int nheads, int tid, int wave, int lane) {
  const int total = nheads << 5;   // nheads * 32
  // --- per-(head,node) source/dest scores ---
  if (tid < 2 * total) {
    int is_d = (tid >= total) ? 1 : 0;
    int p    = tid - is_d * total;
    int head = p >> 5;
    int a    = p & 31;
    const float* av = (is_d ? adv : asv) + head * C1;
    float sum = 0.f;
    if (a < NODES) {
      const half_t* hrow = Hs + a * LDH + head * C1;
      for (int c = 0; c < C1; ++c) sum += (float)hrow[c] * av[c];
    }
    (is_d ? d_sh : s_sh)[p] = sum;
  }
  __syncthreads();
  // --- per-(head,i) leaky-relu + softmax over source nodes j ---
  if (tid < total) {
    int head = tid >> 5;
    int i    = tid & 31;
    half_t* arow = Al + (head * 32 + i) * 32;
    if (i < NODES) {
      float di = d_sh[(head << 5) + i];
      float row[NODES];
      float mx = -1e30f;
      for (int j = 0; j < NODES; ++j) {
        float l = di + s_sh[(head << 5) + j];
        l = (l >= 0.f) ? l : 0.2f * l;
        row[j] = l;
        mx = fmaxf(mx, l);
      }
      float denom = 0.f;
      for (int j = 0; j < NODES; ++j) { row[j] = __expf(row[j] - mx); denom += row[j]; }
      float inv = 1.f / denom;
      for (int j = 0; j < NODES; ++j) arow[j] = (half_t)(row[j] * inv);
      for (int j = NODES; j < 32; ++j) arow[j] = (half_t)0.f;
    } else {
      for (int j = 0; j < 32; ++j) arow[j] = (half_t)0.f;
    }
  }
  __syncthreads();
  // --- aggregation: O = alpha @ H_head + bias, relu (WMMA, K=32 pad) ---
  const int lo = lane & 15;
  const int hi = lane >> 4;
  const int tiles = nheads << 4;   // nheads * 2 mtiles * 8 ntiles
  for (int t = wave; t < tiles; t += 8) {
    int head  = t >> 4;
    int mtile = (t >> 3) & 1;
    int ntile = t & 7;
    const half_t* Arow = Al + (head * 32 + mtile * 16 + lo) * 32;
    const int ncol = head * C1 + ntile * 16 + lo;
    v16h a, b;
#pragma unroll
    for (int i = 0; i < 16; ++i) {
      int k = kidx(i, lane);
      a[i] = Arow[k];
      b[i] = Hs[k * LDH + ncol];
    }
    v8f acc = {};
    acc = __builtin_amdgcn_wmma_f32_16x16x32_f16(
        false, a, false, b, (short)0, acc, false, false);
    float bb = bias[ncol];
#pragma unroll
    for (int v = 0; v < 8; ++v) {
      int m = mtile * 16 + (hi << 3) + v;
      float val = fmaxf(acc[v] + bb, 0.f);
      O[m * LDH + ncol] = (half_t)val;
    }
  }
  __syncthreads();
}

// ---------------- fused 3-layer GAT, one graph per workgroup ---------------
__global__ __launch_bounds__(NTHREADS) void gat_main(
    const float* __restrict__ feat,
    const half_t* __restrict__ W1p, const half_t* __restrict__ W2p,
    const half_t* __restrict__ W3p,
    const float* __restrict__ as1, const float* __restrict__ ad1, const float* __restrict__ b1,
    const float* __restrict__ as2, const float* __restrict__ ad2, const float* __restrict__ b2,
    const float* __restrict__ as3, const float* __restrict__ ad3, const float* __restrict__ b3,
    const float* __restrict__ Wc, const float* __restrict__ bc,
    float* __restrict__ out) {
  extern __shared__ char smem[];
  half_t* X      = (half_t*)smem;                 // [32][512] f16
  half_t* Hs     = X + ROWP * LDH;                // [32][512] f16
  half_t* Al     = Hs + ROWP * LDH;               // [4][32][32] f16 alpha
  float*  s_sh   = (float*)(Al + 4 * 32 * 32);    // [4*32]
  float*  d_sh   = s_sh + 128;                    // [4*32]
  float*  pooled = d_sh + 128;                    // [128]

  const int g    = blockIdx.x;
  const int tid  = threadIdx.x;
  const int wave = tid >> 5;
  const int lane = tid & 31;

  // load features [22][32] -> f16, zero-pad rows 22..31
  for (int idx = tid; idx < ROWP * FIN; idx += NTHREADS) {
    int a = idx >> 5, f = idx & 31;
    X[a * LDH + f] =
        (a < NODES) ? (half_t)feat[((size_t)g * NODES + a) * FIN + f] : (half_t)0.f;
  }
  __syncthreads();

  // layer 1: h = x @ W1 (K=32, N=512), attention (4 heads)
  mm_stage<4>(W1p, X, Hs, FIN, D1, wave, lane);
  __syncthreads();
  gat_attention(Hs, X, Al, s_sh, d_sh, as1, ad1, b1, 4, tid, wave, lane);

  // layer 2: K=512, N=512
  mm_stage<4>(W2p, X, Hs, D1, D1, wave, lane);
  __syncthreads();
  gat_attention(Hs, X, Al, s_sh, d_sh, as2, ad2, b2, 4, tid, wave, lane);

  // layer 3: K=512, N=128, single head (mean over 1 head == identity)
  mm_stage<4>(W3p, X, Hs, D1, C1, wave, lane);
  __syncthreads();
  gat_attention(Hs, X, Al, s_sh, d_sh, as3, ad3, b3, 1, tid, wave, lane);

  // mean pool over the 22 nodes, then final linear [128] -> scalar
  if (tid < C1) {
    float s = 0.f;
    for (int a = 0; a < NODES; ++a) s += (float)X[a * LDH + tid];
    pooled[tid] = s * (1.f / (float)NODES);
  }
  __syncthreads();
  if (tid == 0) {
    float s = bc[0];
    for (int c = 0; c < C1; ++c) s += pooled[c] * Wc[c];
    out[g] = s;
  }
}

extern "C" void kernel_launch(void* const* d_in, const int* in_sizes, int n_in,
                              void* d_out, int out_size, void* d_ws, size_t ws_size,
                              hipStream_t stream) {
  const float* feat = (const float*)d_in[0];
  const float* W1   = (const float*)d_in[1];
  const float* as1  = (const float*)d_in[2];
  const float* ad1  = (const float*)d_in[3];
  const float* b1   = (const float*)d_in[4];
  const float* W2   = (const float*)d_in[5];
  const float* as2  = (const float*)d_in[6];
  const float* ad2  = (const float*)d_in[7];
  const float* b2   = (const float*)d_in[8];
  const float* W3   = (const float*)d_in[9];
  const float* as3  = (const float*)d_in[10];
  const float* ad3  = (const float*)d_in[11];
  const float* b3   = (const float*)d_in[12];
  const float* Wc   = (const float*)d_in[13];
  const float* bc   = (const float*)d_in[14];

  const int G = in_sizes[0] / (NODES * FIN);   // B*T = 4096 graphs

  half_t* wsp = (half_t*)d_ws;                 // ~688 KB of packed f16 weights
  const int total_pack = W1P_ELEMS + W2P_ELEMS + W3P_ELEMS;
  pack_weights<<<(total_pack + NTHREADS - 1) / NTHREADS, NTHREADS, 0, stream>>>(
      W1, W2, W3, wsp);

  const size_t smem_bytes =
      (size_t)(2 * ROWP * LDH + 4 * 32 * 32) * sizeof(half_t) +
      (size_t)(128 + 128 + 128) * sizeof(float);   // 75,264 B

  gat_main<<<G, NTHREADS, smem_bytes, stream>>>(
      feat, wsp, wsp + W1P_ELEMS, wsp + W1P_ELEMS + W2P_ELEMS,
      as1, ad1, b1, as2, ad2, b2, as3, ad3, b3, Wc, bc, (float*)d_out);
}
